// MultiHeadAttention_19473381720265
// MI455X (gfx1250) — matrix operand order repaired and checked by reading, not compiled, not run on hardware
//
#include <hip/hip_runtime.h>
#include <hip/hip_bf16.h>

// ---------------------------------------------------------------------------
// MultiHeadAttention for MI455X (gfx1250, wave32, WMMA bf16 16x16x32)
// B=4, N=2048, D=768, H=16, Dh=48 (padded to 64 in workspace)
//
// Pipeline:
//   1) fp32 -> bf16 casts (x, Wq/Wk/Wv/Wo)
//   2) Q/K proj -> [bh][tok][64] padded; V proj -> TRANSPOSED [bh][64][tok]
//      (one wave computes a full head's 3 N-tiles, reusing the A fragment)
//   3) flash attention: 4 score WMMAs + 3 P@V WMMAs per 32-key chunk,
//      online softmax, P transposed C->A layout via 1KB LDS tile,
//      V^T layout makes every B fragment a single 32B contiguous load
//   4) output proj WMMA GEMM (4 N-tiles/wave) -> fp32 d_out
// ---------------------------------------------------------------------------

typedef __bf16 bf16;
typedef __attribute__((ext_vector_type(16))) __bf16 v16bf;
typedef __attribute__((ext_vector_type(8)))  __bf16 v8bf;
typedef __attribute__((ext_vector_type(8)))  float  v8f;

#define B_   4
#define N_   2048
#define D_   768
#define H_   16
#define DH   48
#define DHP  64
#define M_   (B_ * N_)           // 8192 total tokens

__device__ __forceinline__ bf16 f2bf(float f) {
  unsigned u = __builtin_bit_cast(unsigned, f);
  unsigned r = (u + 0x7FFFu + ((u >> 16) & 1u)) >> 16;   // round-to-nearest-even
  unsigned short s = (unsigned short)r;
  return __builtin_bit_cast(bf16, s);
}

__device__ __forceinline__ v8f vzero8() {
  v8f z;
#pragma unroll
  for (int i = 0; i < 8; ++i) z[i] = 0.0f;
  return z;
}

// Load a 16x32 bf16 A-fragment:
// lane L holds row L%16; hi = L/16 selects contiguous K range {hi*16..+15}.
__device__ __forceinline__ v16bf load_a_frag(const bf16* rowk, int hi) {
  const v8bf a0 = *(const v8bf*)(rowk + hi * 16);
  const v8bf a1 = *(const v8bf*)(rowk + hi * 16 + 8);
  v16bf a;
#pragma unroll
  for (int j = 0; j < 8; ++j) { a[j] = a0[j]; a[8 + j] = a1[j]; }
  return a;
}

// ---------------------------------------------------------------------------
// fp32 -> bf16 conversion (grid-stride)
// ---------------------------------------------------------------------------
__global__ void cvt_f32_bf16(const float* __restrict__ in, bf16* __restrict__ out, int n) {
  int i = blockIdx.x * blockDim.x + threadIdx.x;
  int stride = gridDim.x * blockDim.x;
  for (; i < n; i += stride) out[i] = f2bf(in[i]);
}

// ---------------------------------------------------------------------------
// Q/K/V projection: one wave computes one head's full 16x48 output stripe
// (3 WMMA accumulators sharing one A fragment per K-step).
//   transpose_v == 0 : out[bh][tok][64]  (Q, K; d 48..63 zero-padded)
//   transpose_v == 1 : out[bh][64][tok]  (V^T; rows 48..63 zero-padded)
// grid = (M_/16, H_), block = 32.
// ---------------------------------------------------------------------------
__global__ void proj_qkv(const bf16* __restrict__ X, const bf16* __restrict__ W,
                         bf16* __restrict__ out, int transpose_v) {
  const int lane = threadIdx.x;
  const int lo = lane & 15, hi = lane >> 4;
  const int m0 = blockIdx.x * 16;
  const int h  = blockIdx.y;
  const int n0 = h * DH;

  v8f c[3];
#pragma unroll
  for (int t = 0; t < 3; ++t) c[t] = vzero8();

  const bf16* arow = X + (size_t)(m0 + lo) * D_;
  const bf16* brow0 = W + (size_t)(n0 + lo) * D_;        // B tile t adds t*16 rows
#pragma unroll 4
  for (int k0 = 0; k0 < D_; k0 += 32) {
    v16bf a = load_a_frag(arow + k0, hi);
#pragma unroll
    for (int t = 0; t < 3; ++t) {
      v16bf b = *(const v16bf*)(brow0 + (size_t)t * 16 * D_ + k0 + hi * 16);
      c[t] = __builtin_amdgcn_wmma_f32_16x16x32_bf16(false, a, false, b, (short)0, c[t], false, false);
    }
  }

  const bf16 z = f2bf(0.0f);
  if (!transpose_v) {
#pragma unroll
    for (int r = 0; r < 8; ++r) {
      int m = m0 + r + 8 * hi;
      int b = m >> 11, tok = m & (N_ - 1);
      size_t base = ((size_t)(b * H_ + h) * N_ + tok) * DHP;
#pragma unroll
      for (int t = 0; t < 3; ++t) out[base + t * 16 + lo] = f2bf(c[t][r]);
      out[base + DH + lo] = z;                           // pad d = 48..63
    }
  } else {
#pragma unroll
    for (int r = 0; r < 8; ++r) {
      int m = m0 + r + 8 * hi;
      int b = m >> 11, tok = m & (N_ - 1);
      size_t base = (size_t)(b * H_ + h) * DHP * N_ + tok;
#pragma unroll
      for (int t = 0; t < 3; ++t)
        out[base + (size_t)(t * 16 + lo) * N_] = f2bf(c[t][r]);
      out[base + (size_t)(DH + lo) * N_] = z;            // pad rows d = 48..63
    }
  }
}

// ---------------------------------------------------------------------------
// Flash attention: one wave = one 16-query tile of one (b,h), streaming all
// 2048 keys in chunks of 32. V is pre-transposed so every WMMA operand is a
// contiguous 16/32-byte load.
// grid = (N_/16, B_*H_), block = 32.
// ---------------------------------------------------------------------------
__global__ void attn_kernel(const bf16* __restrict__ Qb, const bf16* __restrict__ Kb,
                            const bf16* __restrict__ Vtb, bf16* __restrict__ attn) {
  __shared__ bf16 Pt[16 * 32];

  const int lane = threadIdx.x;
  const int lo = lane & 15, hi = lane >> 4;
  const int q0 = blockIdx.x * 16;
  const int bh = blockIdx.y;

  const bf16* Q  = Qb  + (size_t)bh * N_ * DHP;
  const bf16* K  = Kb  + (size_t)bh * N_ * DHP;
  const bf16* Vt = Vtb + (size_t)bh * DHP * N_;          // [d][tok]

  // Q A-fragments for the two K=32 chunks of the padded head dim
  const v16bf aq0 = load_a_frag(Q + (size_t)(q0 + lo) * DHP + 0,  hi);
  const v16bf aq1 = load_a_frag(Q + (size_t)(q0 + lo) * DHP + 32, hi);

  float mrow[8], lrow[8];
  v8f acc[3];
#pragma unroll
  for (int r = 0; r < 8; ++r) { mrow[r] = -1e30f; lrow[r] = 0.0f; }
#pragma unroll
  for (int t = 0; t < 3; ++t) acc[t] = vzero8();

  const float sinv = 0.14433756729740643f;               // 1/sqrt(48)

  for (int kc = 0; kc < N_; kc += 32) {
    // prefetch next chunk of K / V^T into cache (global_prefetch_b8)
    if (kc + 32 < N_) {
      __builtin_prefetch(K + (size_t)(kc + 32 + lane) * DHP, 0, 1);
      __builtin_prefetch(Vt + (size_t)lane * N_ + kc + 32, 0, 1);
    }

    // ---- scores: S = Q @ K^T for keys [kc, kc+32) ------------------------
    v8f s0 = vzero8(), s1 = vzero8();
    {
      v16bf bk;
      bk = *(const v16bf*)(K + (size_t)(kc + lo) * DHP + 0  + hi * 16);
      s0 = __builtin_amdgcn_wmma_f32_16x16x32_bf16(false, aq0, false, bk, (short)0, s0, false, false);
      bk = *(const v16bf*)(K + (size_t)(kc + lo) * DHP + 32 + hi * 16);
      s0 = __builtin_amdgcn_wmma_f32_16x16x32_bf16(false, aq1, false, bk, (short)0, s0, false, false);
      bk = *(const v16bf*)(K + (size_t)(kc + 16 + lo) * DHP + 0  + hi * 16);
      s1 = __builtin_amdgcn_wmma_f32_16x16x32_bf16(false, aq0, false, bk, (short)0, s1, false, false);
      bk = *(const v16bf*)(K + (size_t)(kc + 16 + lo) * DHP + 32 + hi * 16);
      s1 = __builtin_amdgcn_wmma_f32_16x16x32_bf16(false, aq1, false, bk, (short)0, s1, false, false);
    }

    // ---- online softmax (rows striped across 16 lanes per half-wave) -----
#pragma unroll
    for (int r = 0; r < 8; ++r) {
      float a = s0[r] * sinv, b = s1[r] * sinv;
      float t = fmaxf(a, b);
      t = fmaxf(t, __shfl_xor(t, 1));
      t = fmaxf(t, __shfl_xor(t, 2));
      t = fmaxf(t, __shfl_xor(t, 4));
      t = fmaxf(t, __shfl_xor(t, 8));
      float mn   = fmaxf(mrow[r], t);
      float corr = __expf(mrow[r] - mn);
      mrow[r] = mn;
      float p0 = __expf(a - mn), p1 = __expf(b - mn);
      float ps = p0 + p1;
      ps += __shfl_xor(ps, 1);
      ps += __shfl_xor(ps, 2);
      ps += __shfl_xor(ps, 4);
      ps += __shfl_xor(ps, 8);
      lrow[r] = lrow[r] * corr + ps;
      acc[0][r] *= corr; acc[1][r] *= corr; acc[2][r] *= corr;
      // stash P in LDS (row-major 16x32) for C-layout -> A-layout transpose
      Pt[(8 * hi + r) * 32 + lo]      = f2bf(p0);
      Pt[(8 * hi + r) * 32 + 16 + lo] = f2bf(p1);
    }
    __syncthreads();

    v16bf pa = load_a_frag(&Pt[lo * 32], hi);            // A-fragment of P (16x32)

    // ---- O += P @ V : B fragments are contiguous rows of V^T -------------
#pragma unroll
    for (int dt = 0; dt < 3; ++dt) {
      v16bf bv = *(const v16bf*)(Vt + (size_t)(dt * 16 + lo) * N_ + kc + hi * 16);
      acc[dt] = __builtin_amdgcn_wmma_f32_16x16x32_bf16(false, pa, false, bv, (short)0, acc[dt], false, false);
    }
    __syncthreads();                                      // Pt reused next chunk
  }

  // ---- normalize and write attn output as bf16 [b, tok, h*48 + d] --------
  const int b = bh >> 4, h = bh & 15;
#pragma unroll
  for (int r = 0; r < 8; ++r) {
    int tok = q0 + r + 8 * hi;
    float inv = 1.0f / lrow[r];
    size_t base = ((size_t)(b * N_ + tok)) * D_ + h * DH;
#pragma unroll
    for (int dt = 0; dt < 3; ++dt)
      attn[base + dt * 16 + lo] = f2bf(acc[dt][r] * inv);
  }
}

// ---------------------------------------------------------------------------
// Output projection: d_out = attn @ Wo.T (fp32). One wave computes a 16x64
// stripe (4 N-tiles sharing one A fragment per K-step).
// grid = (M_/16, D_/64), block = 32.
// ---------------------------------------------------------------------------
__global__ void proj_out(const bf16* __restrict__ X, const bf16* __restrict__ W,
                         float* __restrict__ out) {
  const int lane = threadIdx.x;
  const int lo = lane & 15, hi = lane >> 4;
  const int m0 = blockIdx.x * 16;
  const int n0 = blockIdx.y * 64;

  v8f c[4];
#pragma unroll
  for (int t = 0; t < 4; ++t) c[t] = vzero8();

  const bf16* arow  = X + (size_t)(m0 + lo) * D_;
  const bf16* brow0 = W + (size_t)(n0 + lo) * D_;
#pragma unroll 2
  for (int k0 = 0; k0 < D_; k0 += 32) {
    v16bf a = load_a_frag(arow + k0, hi);
#pragma unroll
    for (int t = 0; t < 4; ++t) {
      v16bf b = *(const v16bf*)(brow0 + (size_t)t * 16 * D_ + k0 + hi * 16);
      c[t] = __builtin_amdgcn_wmma_f32_16x16x32_bf16(false, a, false, b, (short)0, c[t], false, false);
    }
  }
#pragma unroll
  for (int r = 0; r < 8; ++r) {
    size_t base = (size_t)(m0 + r + 8 * hi) * D_ + n0;
#pragma unroll
    for (int t = 0; t < 4; ++t) out[base + t * 16 + lo] = c[t][r];
  }
}

// ---------------------------------------------------------------------------
// Host launch
// ---------------------------------------------------------------------------
extern "C" void kernel_launch(void* const* d_in, const int* in_sizes, int n_in,
                              void* d_out, int out_size, void* d_ws, size_t ws_size,
                              hipStream_t stream) {
  const float* x  = (const float*)d_in[0];
  const float* Wq = (const float*)d_in[1];
  const float* Wk = (const float*)d_in[2];
  const float* Wv = (const float*)d_in[3];
  const float* Wo = (const float*)d_in[4];

  char* ws = (char*)d_ws;
  size_t off = 0;
  auto alloc = [&](size_t bytes) -> char* {
    char* p = ws + off;
    off = (off + bytes + 255) & ~(size_t)255;
    return p;
  };

  bf16* xb  = (bf16*)alloc((size_t)M_ * D_ * sizeof(bf16));
  bf16* wqb = (bf16*)alloc((size_t)D_ * D_ * sizeof(bf16));
  bf16* wkb = (bf16*)alloc((size_t)D_ * D_ * sizeof(bf16));
  bf16* wvb = (bf16*)alloc((size_t)D_ * D_ * sizeof(bf16));
  bf16* wob = (bf16*)alloc((size_t)D_ * D_ * sizeof(bf16));
  bf16* qb  = (bf16*)alloc((size_t)B_ * H_ * N_ * DHP * sizeof(bf16));
  bf16* kb  = (bf16*)alloc((size_t)B_ * H_ * N_ * DHP * sizeof(bf16));
  bf16* vtb = (bf16*)alloc((size_t)B_ * H_ * DHP * N_ * sizeof(bf16));
  bf16* ab  = (bf16*)alloc((size_t)M_ * D_ * sizeof(bf16));

  // fp32 -> bf16 conversions
  cvt_f32_bf16<<<2048, 256, 0, stream>>>(x,  xb,  M_ * D_);
  cvt_f32_bf16<<<512,  256, 0, stream>>>(Wq, wqb, D_ * D_);
  cvt_f32_bf16<<<512,  256, 0, stream>>>(Wk, wkb, D_ * D_);
  cvt_f32_bf16<<<512,  256, 0, stream>>>(Wv, wvb, D_ * D_);
  cvt_f32_bf16<<<512,  256, 0, stream>>>(Wo, wob, D_ * D_);

  // projections: Q/K padded [bh][tok][64]; V transposed [bh][64][tok]
  dim3 gp(M_ / 16, H_);
  proj_qkv<<<gp, 32, 0, stream>>>(xb, wqb, qb,  0);
  proj_qkv<<<gp, 32, 0, stream>>>(xb, wkb, kb,  0);
  proj_qkv<<<gp, 32, 0, stream>>>(xb, wvb, vtb, 1);

  // flash attention
  dim3 ga(N_ / 16, B_ * H_);
  attn_kernel<<<ga, 32, 0, stream>>>(qb, kb, vtb, ab);

  // output projection -> fp32
  dim3 go(M_ / 16, D_ / 64);
  proj_out<<<go, 32, 0, stream>>>(ab, wob, (float*)d_out);
}
